// HierarchicalGRU_4535485464793
// MI455X (gfx1250) — matrix operand order repaired and checked by reading, not compiled, not run on hardware
//
#include <hip/hip_runtime.h>
#include <hip/hip_bf16.h>

// ---------------------------------------------------------------------------
// HierarchicalGRU for MI455X (gfx1250), wave32 + v_wmma_f32_16x16x32_f16
// B=16, U=48, W=96, V=30000, D=256, H=512
// ---------------------------------------------------------------------------

#define Bq   16
#define Uq   48
#define Wq   96
#define Dq   256
#define Hq   512
#define G3H  1536           // 3*H
#define NSEQ 768            // B*U

typedef __attribute__((ext_vector_type(16))) _Float16 v16h;
typedef __attribute__((ext_vector_type(8)))  float    v8f;

__device__ __forceinline__ float sigm(float x) { return 1.0f / (1.0f + __expf(-x)); }

__device__ __forceinline__ v8f wmma16(v16h a, v16h b, v8f c) {
  // D = A(16x32 f16) * B(32x16 f16) + C(16x16 f32)
  return __builtin_amdgcn_wmma_f32_16x16x32_f16(
      /*neg_a=*/false, a, /*neg_b=*/false, b,
      /*c_mod=*/(short)0, c, /*reuse_a=*/false, /*reuse_b=*/false);
}

// A-matrix fragment (16x32 f16, MxK) built from an LDS tile.
// Lane L: m = L&15, K-half = (L>>4)*8.  Element e (0..15): rel_k = 8*half + e + (e&8).
__device__ __forceinline__ v16h load_a_frag(const _Float16* base, int stride, int lane, int kbase) {
  const _Float16* r = base + (lane & 15) * stride + kbase + ((lane >> 4) << 3);
  v16h a;
#pragma unroll
  for (int e = 0; e < 16; e += 2) {
    int rk = e + (e & 8);
    a[e]     = r[rk];
    a[e + 1] = r[rk + 1];
  }
  return a;
}

// B-matrix fragment (32x16 f16, KxN), pre-swizzled in global memory:
// frag layout: [fragIdx][lane][16 f16], so this is one contiguous 32B vector load.
__device__ __forceinline__ v16h load_b_frag(const _Float16* __restrict__ frags, int fragIdx, int lane) {
  return *reinterpret_cast<const v16h*>(frags + (size_t)fragIdx * 512 + lane * 16);
}

// ---------------------------------------------------------------------------
// Weight pack: w (1536 x K, row-major f32) -> WMMA B fragments (f16).
// Fragment f = nt*KT + kt.  Element (lane, e): n = nt*16 + (lane&15),
// k = kt*32 + (lane>>4)*8 + e + (e&8).
// ---------------------------------------------------------------------------
__global__ void pack_b_frags(const float* __restrict__ w, _Float16* __restrict__ frags, int K) {
  int KT = K >> 5;
  int total = 96 * KT * 512;
  for (int idx = blockIdx.x * blockDim.x + threadIdx.x; idx < total; idx += gridDim.x * blockDim.x) {
    int f    = idx >> 9;
    int r    = idx & 511;
    int lane = r >> 4;
    int e    = r & 15;
    int nt   = f / KT;
    int kt   = f - nt * KT;
    int n    = (nt << 4) + (lane & 15);
    int k    = (kt << 5) + ((lane >> 4) << 3) + e + (e & 8);
    frags[idx] = (_Float16)w[n * K + k];
  }
}

__global__ void zero_f32(float* __restrict__ p, int n) {
  for (int i = blockIdx.x * blockDim.x + threadIdx.x; i < n; i += gridDim.x * blockDim.x) p[i] = 0.0f;
}

__global__ void bcast_h0(const float* __restrict__ h0, float* __restrict__ h) {
  h[(blockIdx.x << 9) + threadIdx.x] = h0[threadIdx.x];   // 16 blocks x 512 threads
}

// ---------------------------------------------------------------------------
// Word-level GRU, one timestep.  Grid: (48 M-tiles, 8 N-groups), block 128 (4 waves).
// Per wave: one 16x16 h_new tile, 6 accumulators, 72 WMMAs (3x8 input-path + 3x16 hidden-path).
// x_t gathered from the embedding table inline (no embed materialization).
// ---------------------------------------------------------------------------
#define WLDS 770            // 768 cols + 2 pad -> 385 dwords/row, conflict-free
__global__ __launch_bounds__(128) void word_step(
    const int* __restrict__ input, const float* __restrict__ emb, const int* __restrict__ w_len,
    const _Float16* __restrict__ wih_frag, const _Float16* __restrict__ whh_frag,
    const float* __restrict__ b_ih, const float* __restrict__ b_hh,
    const float* __restrict__ h_prev, float* __restrict__ h_next,
    float* __restrict__ utt_input, int t) {
  __shared__ _Float16 lds[16 * WLDS];
  __shared__ int ids[16];
  const int tid   = threadIdx.x;
  const int mtile = blockIdx.x;

  if (tid < 16) ids[tid] = input[(mtile * 16 + tid) * Wq + t];
  __syncthreads();

  for (int e = tid; e < 16 * 768; e += 128) {
    int row = e / 768;
    int col = e - row * 768;
    float v;
    if (col < Dq) v = emb[ids[row] * Dq + col];                              // x_t
    else          v = h_prev[(mtile * 16 + row) * Hq + (col - Dq)];          // h_{t-1}
    lds[row * WLDS + col] = (_Float16)v;
  }
  __syncthreads();

  const int wave  = tid >> 5;
  const int lane  = tid & 31;
  const int ntile = (blockIdx.y << 2) + wave;          // 0..31 over H

  v8f ai0 = {}, ai1 = {}, ai2 = {}, ah0 = {}, ah1 = {}, ah2 = {};
#pragma unroll 4
  for (int kt = 0; kt < 8; ++kt) {                     // input path, K = 256
    v16h a  = load_a_frag(lds, WLDS, lane, kt * 32);
    ai0 = wmma16(a, load_b_frag(wih_frag, (ntile)      * 8 + kt, lane), ai0);
    ai1 = wmma16(a, load_b_frag(wih_frag, (32 + ntile) * 8 + kt, lane), ai1);
    ai2 = wmma16(a, load_b_frag(wih_frag, (64 + ntile) * 8 + kt, lane), ai2);
  }
#pragma unroll 4
  for (int kt = 0; kt < 16; ++kt) {                    // hidden path, K = 512
    v16h a  = load_a_frag(lds + Dq, WLDS, lane, kt * 32);
    ah0 = wmma16(a, load_b_frag(whh_frag, (ntile)      * 16 + kt, lane), ah0);
    ah1 = wmma16(a, load_b_frag(whh_frag, (32 + ntile) * 16 + kt, lane), ah1);
    ah2 = wmma16(a, load_b_frag(whh_frag, (64 + ntile) * 16 + kt, lane), ah2);
  }

  const int half = lane >> 4;
  const int ng   = (ntile << 4) + (lane & 15);
  const float bri = b_ih[ng],        brh = b_hh[ng];
  const float bzi = b_ih[Hq + ng],   bzh = b_hh[Hq + ng];
  const float bni = b_ih[2*Hq + ng], bnh = b_hh[2*Hq + ng];
#pragma unroll
  for (int p = 0; p < 8; ++p) {                        // C layout: m = p + 8*half, n = lane&15
    int seq   = mtile * 16 + (half << 3) + p;
    float r   = sigm(ai0[p] + bri + ah0[p] + brh);
    float z   = sigm(ai1[p] + bzi + ah1[p] + bzh);
    float nn  = tanhf(ai2[p] + bni + r * (ah2[p] + bnh));
    float ho  = h_prev[seq * Hq + ng];
    float hn  = (1.0f - z) * nn + z * ho;
    h_next[seq * Hq + ng] = hn;
    if (w_len[seq] == t + 1) utt_input[seq * Hq + ng] = hn;   // capture at wl-1
  }
}

// ---------------------------------------------------------------------------
// Generic 16-row GRU step (utterance + segment levels).  Grid: 8 blocks x 4 waves.
// K = 512 (x) + 512 (h).  Writes h_new to hnext and (optionally) outbase.
// ---------------------------------------------------------------------------
#define SLDS 1026           // 1024 cols + 2 pad -> 513 dwords/row
__global__ __launch_bounds__(128) void gru16_step(
    const float* __restrict__ xbase, int xstride,
    const float* __restrict__ hprev,
    const _Float16* __restrict__ wih_frag, const _Float16* __restrict__ whh_frag,
    const float* __restrict__ b_ih, const float* __restrict__ b_hh,
    float* __restrict__ hnext, float* __restrict__ outbase, int ostride) {
  __shared__ _Float16 lds[16 * SLDS];
  const int tid = threadIdx.x;
  for (int e = tid; e < 16 * 1024; e += 128) {
    int row = e >> 10;
    int col = e & 1023;
    float v = (col < Hq) ? xbase[row * xstride + col] : hprev[(row << 9) + (col - Hq)];
    lds[row * SLDS + col] = (_Float16)v;
  }
  __syncthreads();

  const int wave  = tid >> 5;
  const int lane  = tid & 31;
  const int ntile = (blockIdx.x << 2) + wave;

  v8f ai0 = {}, ai1 = {}, ai2 = {}, ah0 = {}, ah1 = {}, ah2 = {};
#pragma unroll 4
  for (int kt = 0; kt < 16; ++kt) {
    v16h a  = load_a_frag(lds, SLDS, lane, kt * 32);
    ai0 = wmma16(a, load_b_frag(wih_frag, (ntile)      * 16 + kt, lane), ai0);
    ai1 = wmma16(a, load_b_frag(wih_frag, (32 + ntile) * 16 + kt, lane), ai1);
    ai2 = wmma16(a, load_b_frag(wih_frag, (64 + ntile) * 16 + kt, lane), ai2);
  }
#pragma unroll 4
  for (int kt = 0; kt < 16; ++kt) {
    v16h a  = load_a_frag(lds + Hq, SLDS, lane, kt * 32);
    ah0 = wmma16(a, load_b_frag(whh_frag, (ntile)      * 16 + kt, lane), ah0);
    ah1 = wmma16(a, load_b_frag(whh_frag, (32 + ntile) * 16 + kt, lane), ah1);
    ah2 = wmma16(a, load_b_frag(whh_frag, (64 + ntile) * 16 + kt, lane), ah2);
  }

  const int half = lane >> 4;
  const int ng   = (ntile << 4) + (lane & 15);
  const float bri = b_ih[ng],        brh = b_hh[ng];
  const float bzi = b_ih[Hq + ng],   bzh = b_hh[Hq + ng];
  const float bni = b_ih[2*Hq + ng], bnh = b_hh[2*Hq + ng];
#pragma unroll
  for (int p = 0; p < 8; ++p) {
    int m     = (half << 3) + p;                       // batch row 0..15
    float r   = sigm(ai0[p] + bri + ah0[p] + brh);
    float z   = sigm(ai1[p] + bzi + ah1[p] + bzh);
    float nn  = tanhf(ai2[p] + bni + r * (ah2[p] + bnh));
    float ho  = hprev[(m << 9) + ng];
    float hn  = (1.0f - z) * nn + z * ho;
    hnext[(m << 9) + ng] = hn;
    if (outbase) outbase[m * ostride + ng] = hn;
  }
}

// ---------------------------------------------------------------------------
// Adaptive gate: g = sigmoid(h_new @ w_g + b_g); boundary reset to h0.
// Grid: 16 blocks (one per batch row) x 512 threads.
// ---------------------------------------------------------------------------
__global__ void utt_gate(const float* __restrict__ hnew, const float* __restrict__ w_g,
                         const float* __restrict__ b_g, const float* __restrict__ h0,
                         float* __restrict__ hnext, int* __restrict__ boundary, int u) {
  __shared__ float red[512];
  __shared__ int bsh;
  const int b = blockIdx.x, j = threadIdx.x;
  float hv = hnew[(b << 9) + j];
  red[j] = hv * w_g[j];
  __syncthreads();
  for (int s = 256; s > 0; s >>= 1) {
    if (j < s) red[j] += red[j + s];
    __syncthreads();
  }
  if (j == 0) {
    float g = sigm(red[0] + b_g[0]);
    int bd  = (g > 0.5f) ? 1 : 0;
    boundary[b * Uq + u] = bd;
    bsh = bd;
  }
  __syncthreads();
  hnext[(b << 9) + j] = bsh ? h0[j] : hv;
}

// ---------------------------------------------------------------------------
// Segment compaction + s_len.  Grid: 16 blocks x 512 threads (one block per b).
// ---------------------------------------------------------------------------
__global__ void seg_prep(const float* __restrict__ utt_output, const int* __restrict__ boundary,
                         const int* __restrict__ u_len, float* __restrict__ seg_input,
                         float* __restrict__ s_len_out) {
  const int b = blockIdx.x, j = threadIdx.x;
  for (int u = 0; u < Uq; ++u) seg_input[(b * Uq + u) * Hq + j] = 0.0f;
  const int ul = u_len[b];
  int cnt = 0;
  for (int u = 0; u < Uq; ++u) {
    bool valid = (u < ul);
    bool mask  = ((boundary[b * Uq + u] != 0) || (u == ul - 1)) && valid;
    if (mask) {
      seg_input[(b * Uq + cnt) * Hq + j] = utt_output[(b * Uq + u) * Hq + j];
      ++cnt;
    }
  }
  if (j == 0) s_len_out[b] = (float)cnt;
}

// ---------------------------------------------------------------------------
// Host-side orchestration (graph-capture safe: only kernel launches on stream)
// ---------------------------------------------------------------------------
extern "C" void kernel_launch(void* const* d_in, const int* in_sizes, int n_in,
                              void* d_out, int out_size, void* d_ws, size_t ws_size,
                              hipStream_t stream) {
  const int*   input  = (const int*)  d_in[0];
  const int*   u_len  = (const int*)  d_in[1];
  const int*   w_len  = (const int*)  d_in[2];
  const float* emb    = (const float*)d_in[3];
  const float* w_ih_w = (const float*)d_in[4];
  const float* w_hh_w = (const float*)d_in[5];
  const float* b_ih_w = (const float*)d_in[6];
  const float* b_hh_w = (const float*)d_in[7];
  const float* w_ih_u = (const float*)d_in[8];
  const float* w_hh_u = (const float*)d_in[9];
  const float* b_ih_u = (const float*)d_in[10];
  const float* b_hh_u = (const float*)d_in[11];
  const float* w_g    = (const float*)d_in[12];
  const float* b_g    = (const float*)d_in[13];
  const float* w_ih_s = (const float*)d_in[14];
  const float* w_hh_s = (const float*)d_in[15];
  const float* b_ih_s = (const float*)d_in[16];
  const float* b_hh_s = (const float*)d_in[17];
  const float* h0     = (const float*)d_in[18];

  // ---- workspace layout (all 256B aligned) ----
  char* ws = (char*)d_ws;
  size_t off = 0;
  auto alloc = [&](size_t bytes) { char* p = ws + off; off = (off + bytes + 255) & ~(size_t)255; return p; };
  _Float16* fWihW = (_Float16*)alloc(96 *  8 * 512 * sizeof(_Float16));   // (1536,256)
  _Float16* fWhhW = (_Float16*)alloc(96 * 16 * 512 * sizeof(_Float16));   // (1536,512)
  _Float16* fWihU = (_Float16*)alloc(96 * 16 * 512 * sizeof(_Float16));
  _Float16* fWhhU = (_Float16*)alloc(96 * 16 * 512 * sizeof(_Float16));
  _Float16* fWihS = (_Float16*)alloc(96 * 16 * 512 * sizeof(_Float16));
  _Float16* fWhhS = (_Float16*)alloc(96 * 16 * 512 * sizeof(_Float16));
  float* h_w      = (float*)alloc(2 * NSEQ * Hq * sizeof(float));         // double-buffered word h
  float* utt_in   = (float*)alloc(NSEQ * Hq * sizeof(float));             // (B,U,H)
  float* utt_h    = (float*)alloc(2 * Bq * Hq * sizeof(float));
  float* utt_hnew = (float*)alloc(Bq * Hq * sizeof(float));
  float* utt_out  = (float*)alloc(Bq * Uq * Hq * sizeof(float));
  int*   bnd      = (int*)  alloc(Bq * Uq * sizeof(int));
  float* seg_in   = (float*)alloc(Bq * Uq * Hq * sizeof(float));
  float* seg_h    = (float*)alloc(2 * Bq * Hq * sizeof(float));

  float* out      = (float*)d_out;                // [0, B*U*H) = s_output, then B s_len values

  // ---- pack weights into WMMA B fragments (f16) ----
  pack_b_frags<<<768, 256, 0, stream>>>(w_ih_w, fWihW, Dq);
  pack_b_frags<<<768, 256, 0, stream>>>(w_hh_w, fWhhW, Hq);
  pack_b_frags<<<768, 256, 0, stream>>>(w_ih_u, fWihU, Hq);
  pack_b_frags<<<768, 256, 0, stream>>>(w_hh_u, fWhhU, Hq);
  pack_b_frags<<<768, 256, 0, stream>>>(w_ih_s, fWihS, Hq);
  pack_b_frags<<<768, 256, 0, stream>>>(w_hh_s, fWhhS, Hq);

  // ---- init hidden states ----
  zero_f32<<<768, 256, 0, stream>>>(h_w, NSEQ * Hq);          // word h0 = 0 (buffer 0)
  bcast_h0<<<Bq, Hq, 0, stream>>>(h0, utt_h);                 // utterance h0 broadcast
  zero_f32<<<32, 256, 0, stream>>>(seg_h, Bq * Hq);           // segment h0 = 0

  // ---- word-level GRU: 96 sequential steps ----
  for (int t = 0; t < Wq; ++t) {
    float* hp = h_w + (size_t)(t & 1) * NSEQ * Hq;
    float* hn = h_w + (size_t)((t + 1) & 1) * NSEQ * Hq;
    word_step<<<dim3(48, 8), 128, 0, stream>>>(input, emb, w_len, fWihW, fWhhW,
                                               b_ih_w, b_hh_w, hp, hn, utt_in, t);
  }

  // ---- utterance-level adaptive GRU: 48 sequential steps ----
  for (int u = 0; u < Uq; ++u) {
    float* hp = utt_h + (size_t)(u & 1) * Bq * Hq;
    float* hn = utt_h + (size_t)((u + 1) & 1) * Bq * Hq;
    gru16_step<<<8, 128, 0, stream>>>(utt_in + u * Hq, Uq * Hq, hp, fWihU, fWhhU,
                                      b_ih_u, b_hh_u, utt_hnew, utt_out + u * Hq, Uq * Hq);
    utt_gate<<<Bq, Hq, 0, stream>>>(utt_hnew, w_g, b_g, h0, hn, bnd, u);
  }

  // ---- segment compaction ----
  seg_prep<<<Bq, Hq, 0, stream>>>(utt_out, bnd, u_len, seg_in, out + Bq * Uq * Hq);

  // ---- segment-level GRU: 48 sequential steps, writes s_output straight to d_out ----
  for (int u = 0; u < Uq; ++u) {
    float* hp = seg_h + (size_t)(u & 1) * Bq * Hq;
    float* hn = seg_h + (size_t)((u + 1) & 1) * Bq * Hq;
    gru16_step<<<8, 128, 0, stream>>>(seg_in + u * Hq, Uq * Hq, hp, fWihS, fWhhS,
                                      b_ih_s, b_hh_s, hn, out + u * Hq, Uq * Hq);
  }
}
